// Bert_33354716021492
// MI455X (gfx1250) — matrix operand order repaired and checked by reading, not compiled, not run on hardware
//
#include <hip/hip_runtime.h>
#include <cstdint>
#include <cstddef>

typedef __attribute__((ext_vector_type(16))) _Float16 v16h;
typedef __attribute__((ext_vector_type(8)))  float    v8f;
typedef unsigned int uint32x4 __attribute__((ext_vector_type(4)));
typedef int          int32x4  __attribute__((ext_vector_type(4)));
typedef int          int32x8  __attribute__((ext_vector_type(8)));

union Frag { v16h h; uint4 q[2]; };
union Acc  { v8f  v; float f[8]; };

#define TB 256

// ---------------------------------------------------------------------------
// TDM issue: 2D f16 tile (tile_w elems x tile_h rows) global -> LDS, with LDS
// row padding.  D# packed per CDNA5 ISA (group0/group1; 2D so
// groups 2/3 zero).  data_size=2B; pad codes: interval 4 -> 32 DWORDs (64
// halves), amount 3 -> 4 DWORDs (8 halves) => 72-half LDS row stride.
// ---------------------------------------------------------------------------
__device__ inline void tdm_load_tile_f16(unsigned int lds_byte_addr,
                                         const _Float16* gsrc,
                                         unsigned int tile_w, unsigned int tile_h,
                                         long stride_elems,
                                         unsigned int pad_interval_code,
                                         unsigned int pad_amount_code,
                                         long rem0, long rem1)
{
    unsigned long long ga = (unsigned long long)(uintptr_t)gsrc;
    uint32x4 g0;
    g0.x = 1u;                                                   // count=1, user mode
    g0.y = lds_byte_addr;                                        // LDS dest
    g0.z = (unsigned int)ga;                                     // global_addr[31:0]
    g0.w = (unsigned int)((ga >> 32) & 0x01FFFFFFu) | (2u << 30);// addr[56:32] | type=2
    unsigned long long flags =
          (1ull << 16)                                           // data_size = 2 bytes
        | (1ull << 20)                                           // pad_enable
        | ((unsigned long long)pad_interval_code << 22)
        | ((unsigned long long)pad_amount_code   << 25);
    unsigned long long td0 = (unsigned long long)rem0;           // tensor_dim0 (bits 79:48)
    unsigned long long td1 = (unsigned long long)rem1;           // tensor_dim1 (bits 111:80)
    unsigned long long st0 = (unsigned long long)stride_elems;   // dim0 stride (bits 207:160)
    unsigned long long q0 = flags | ((td0 & 0xFFFFull) << 48);
    unsigned long long q1 = (td0 >> 16) | (td1 << 16) |
                            (((unsigned long long)tile_w & 0xFFFFull) << 48); // tile_dim0 127:112
    unsigned long long q2 = ((unsigned long long)tile_h & 0xFFFFull)          // tile_dim1 143:128
                            | ((st0 & 0xFFFFFFFFull) << 32);
    unsigned long long q3 = (st0 >> 32) & 0xFFFFull;
    int32x8 g1;
    g1[0] = (int)(unsigned int)q0;  g1[1] = (int)(unsigned int)(q0 >> 32);
    g1[2] = (int)(unsigned int)q1;  g1[3] = (int)(unsigned int)(q1 >> 32);
    g1[4] = (int)(unsigned int)q2;  g1[5] = (int)(unsigned int)(q2 >> 32);
    g1[6] = (int)(unsigned int)q3;  g1[7] = (int)(unsigned int)(q3 >> 32);
    int32x4 z4 = {};
#if defined(__clang_major__) && __clang_major__ >= 23
    int32x8 z8 = {};
    __builtin_amdgcn_tensor_load_to_lds(g0, g1, z4, z4, z8, 0);
#else
    __builtin_amdgcn_tensor_load_to_lds(g0, g1, z4, z4, 0);
#endif
}

__device__ inline unsigned int lds_addr_of(const void* p)
{
    // flat address in the LDS aperture: the low 32 bits are the LDS byte offset (ISA §10.2)
    return (unsigned int)(uintptr_t)p;
}

// ---------------------------------------------------------------------------
// TDM-fed NT GEMM: C[M,N] = act(alpha*A*Bt^T + bias).  128x64 block tile,
// BK=64, 8 wave32s as 4x2; each wave owns a 32x32 slab (4 accums, 8 WMMAs per
// K-step).  Wave0 DMAs the A tile, wave1 the B tile, double-buffered with
// s_wait_tensorcnt.  M%128==0, N%64==0, K%64==0 required.
// ---------------------------------------------------------------------------
#define BM 128
#define BN 64
#define BKT 64
#define LDK 72   // 64 + 8 halves pad (matches TDM pad codes 4/3)

__global__ __launch_bounds__(TB)
void gemm_wmma_nt_tdm(const _Float16* __restrict__ A, const _Float16* __restrict__ B,
                      const float* __restrict__ bias,
                      float* __restrict__ Cf, _Float16* __restrict__ Ch,
                      int M, int N, int K, int lda, int ldb, int ldc,
                      float alpha, int act, int outTrans)
{
    __shared__ __align__(16) _Float16 lA[2][BM][LDK];
    __shared__ __align__(16) _Float16 lB[2][BN][LDK];

    int tid  = threadIdx.x;
    int lane = tid & 31;
    int w    = tid >> 5;
    int wm   = w >> 1;              // 0..3 row slab (32 rows each)
    int wn   = w & 1;               // 0..1 col slab (32 cols each)
    int m16  = lane & 15;
    int hi   = lane >> 4;

    long rowBase = (long)blockIdx.y * BM;
    long colBase = (long)blockIdx.x * BN;

    Acc c00, c01, c10, c11;
#pragma unroll
    for (int i = 0; i < 8; ++i) { c00.f[i] = 0.f; c01.f[i] = 0.f; c10.f[i] = 0.f; c11.f[i] = 0.f; }

    const int nk = K / BKT;
    if (w == 0)
        tdm_load_tile_f16(lds_addr_of(&lA[0][0][0]), A + rowBase * (long)lda,
                          BKT, BM, lda, 4, 3, K, M - rowBase);
    if (w == 1)
        tdm_load_tile_f16(lds_addr_of(&lB[0][0][0]), B + colBase * (long)ldb,
                          BKT, BN, ldb, 4, 3, K, N - colBase);

    for (int s = 0; s < nk; ++s) {
        int buf = s & 1;
        if (s + 1 < nk) {
            long k1 = (long)(s + 1) * BKT;
            if (w == 0)
                tdm_load_tile_f16(lds_addr_of(&lA[buf ^ 1][0][0]),
                                  A + rowBase * (long)lda + k1,
                                  BKT, BM, lda, 4, 3, K - k1, M - rowBase);
            if (w == 1)
                tdm_load_tile_f16(lds_addr_of(&lB[buf ^ 1][0][0]),
                                  B + colBase * (long)ldb + k1,
                                  BKT, BN, ldb, 4, 3, K - k1, N - colBase);
            if (w < 2) __builtin_amdgcn_s_wait_tensorcnt(1);   // current tile done
        } else {
            if (w < 2) __builtin_amdgcn_s_wait_tensorcnt(0);
        }
        __syncthreads();

#pragma unroll
        for (int ks = 0; ks < BKT; ks += 32) {
            Frag aLo, aHi, fb0, fb1;
            int am = wm * 32 + m16;
            aLo.q[0] = *(const uint4*)&lA[buf][am][ks + 16 * hi];
            aLo.q[1] = *(const uint4*)&lA[buf][am][ks + 16 * hi + 8];
            aHi.q[0] = *(const uint4*)&lA[buf][am + 16][ks + 16 * hi];
            aHi.q[1] = *(const uint4*)&lA[buf][am + 16][ks + 16 * hi + 8];
            int bn = wn * 32 + m16;
            fb0.q[0] = *(const uint4*)&lB[buf][bn][ks + 16 * hi];
            fb0.q[1] = *(const uint4*)&lB[buf][bn][ks + 16 * hi + 8];
            fb1.q[0] = *(const uint4*)&lB[buf][bn + 16][ks + 16 * hi];
            fb1.q[1] = *(const uint4*)&lB[buf][bn + 16][ks + 16 * hi + 8];
            c00.v = __builtin_amdgcn_wmma_f32_16x16x32_f16(false, aLo.h, false, fb0.h, (short)0, c00.v, false, false);
            c01.v = __builtin_amdgcn_wmma_f32_16x16x32_f16(false, aLo.h, false, fb1.h, (short)0, c01.v, false, false);
            c10.v = __builtin_amdgcn_wmma_f32_16x16x32_f16(false, aHi.h, false, fb0.h, (short)0, c10.v, false, false);
            c11.v = __builtin_amdgcn_wmma_f32_16x16x32_f16(false, aHi.h, false, fb1.h, (short)0, c11.v, false, false);
        }
        __syncthreads();
    }

    auto store_tile = [&](Acc& a, long gm0, long gn0) {
#pragma unroll
        for (int r = 0; r < 8; ++r) {
            long gm = gm0 + r + 8 * hi;
            long gn = gn0 + m16;
            float v = a.f[r] * alpha;
            if (bias) v += bias[gn];
            if (act == 1) v = 0.5f * v * (1.0f + erff(v * 0.70710678f));
            if (Ch) {
                long idx = outTrans ? (gn * (long)ldc + gm) : (gm * (long)ldc + gn);
                Ch[idx] = (_Float16)v;
            } else {
                Cf[gm * (long)ldc + gn] = v;
            }
        }
    };
    store_tile(c00, rowBase + wm * 32,      colBase + wn * 32);
    store_tile(c01, rowBase + wm * 32,      colBase + wn * 32 + 16);
    store_tile(c10, rowBase + wm * 32 + 16, colBase + wn * 32);
    store_tile(c11, rowBase + wm * 32 + 16, colBase + wn * 32 + 16);
}

// ---------------------------------------------------------------------------
// Round-1 NT GEMM (VMEM->LDS path), kept for the batched attention
// GEMMs.  64x64 tile, BK=32, 8 waves x (16x32 slab).
// ---------------------------------------------------------------------------
__global__ __launch_bounds__(TB)
void gemm_wmma_nt(const _Float16* __restrict__ A, const _Float16* __restrict__ B,
                  const float* __restrict__ bias,
                  float* __restrict__ Cf, _Float16* __restrict__ Ch,
                  int M, int N, int K, int lda, int ldb, int ldc,
                  long sAb, long sAh, long sBb, long sBh, long sCb, long sCh,
                  int nh, float alpha, int act, int outTrans)
{
    int z  = blockIdx.z;
    int bb = z / nh, hh = z - bb * nh;
    A += (long)bb * sAb + (long)hh * sAh;
    B += (long)bb * sBb + (long)hh * sBh;
    long cOff = (long)bb * sCb + (long)hh * sCh;

    __shared__ __align__(16) _Float16 lA[64][40];
    __shared__ __align__(16) _Float16 lB[64][40];

    int tid  = threadIdx.x;
    int lane = tid & 31;
    int w    = tid >> 5;
    int wm   = w >> 1;
    int wn   = w & 1;
    int m16  = lane & 15;
    int hi   = lane >> 4;

    long rowBase = (long)blockIdx.y * 64;
    long colBase = (long)blockIdx.x * 64;

    int ldr  = tid >> 2;
    int ldc8 = (tid & 3) * 8;

    const _Float16* aSrc = A + (rowBase + ldr) * (long)lda + ldc8;
    const _Float16* bSrc = B + (colBase + ldr) * (long)ldb + ldc8;

    Acc a0, a1;
#pragma unroll
    for (int i = 0; i < 8; ++i) { a0.f[i] = 0.f; a1.f[i] = 0.f; }

    for (int k0 = 0; k0 < K; k0 += 32) {
        uint4 va = *(const uint4*)(aSrc);
        uint4 vb = *(const uint4*)(bSrc);
        aSrc += 32; bSrc += 32;
        if (k0 + 32 < K) {
            __builtin_prefetch(aSrc, 0, 1);
            __builtin_prefetch(bSrc, 0, 1);
        }
        *(uint4*)&lA[ldr][ldc8] = va;
        *(uint4*)&lB[ldr][ldc8] = vb;
        __syncthreads();

        Frag fa, fb0, fb1;
        int am = wm * 16 + m16;
        fa.q[0] = *(const uint4*)&lA[am][16 * hi];
        fa.q[1] = *(const uint4*)&lA[am][16 * hi + 8];
        int bn = wn * 32 + m16;
        fb0.q[0] = *(const uint4*)&lB[bn][16 * hi];
        fb0.q[1] = *(const uint4*)&lB[bn][16 * hi + 8];
        fb1.q[0] = *(const uint4*)&lB[bn + 16][16 * hi];
        fb1.q[1] = *(const uint4*)&lB[bn + 16][16 * hi + 8];

        a0.v = __builtin_amdgcn_wmma_f32_16x16x32_f16(false, fa.h, false, fb0.h,
                                                      (short)0, a0.v, false, false);
        a1.v = __builtin_amdgcn_wmma_f32_16x16x32_f16(false, fa.h, false, fb1.h,
                                                      (short)0, a1.v, false, false);
        __syncthreads();
    }

#pragma unroll
    for (int r = 0; r < 8; ++r) {
        long gm = rowBase + wm * 16 + r + 8 * hi;
#pragma unroll
        for (int t = 0; t < 2; ++t) {
            long gn = colBase + wn * 32 + t * 16 + m16;
            float v = (t ? a1.f[r] : a0.f[r]) * alpha;
            if (bias) v += bias[gn];
            if (act == 1) v = 0.5f * v * (1.0f + erff(v * 0.70710678f));
            if (Ch) {
                long idx = outTrans ? (cOff + gn * (long)ldc + gm)
                                    : (cOff + gm * (long)ldc + gn);
                Ch[idx] = (_Float16)v;
            } else {
                Cf[cOff + gm * (long)ldc + gn] = v;
            }
        }
    }
}

// ---------------------------------------------------------------------------
__global__ __launch_bounds__(TB)
void embed_kernel(const int* __restrict__ x, const float* __restrict__ tok,
                  const float* __restrict__ seg, float* __restrict__ out)
{
    int row = blockIdx.x;
    int t   = row & 511;
    int token = x[row];
    int sg    = (t >= 257) ? 1 : 0;
    const float* tp = tok + (long)token * 768;
    const float* sp = seg + (long)sg * 768;
    for (int d = threadIdx.x; d < 768; d += TB) {
        float dd = (2.0f * d) / 768.0f;
        float p  = (float)t * expf(-dd * 9.210340371976184f);
        float pe = ((d & 1) == 0) ? sinf(p) : cosf(p);
        out[(long)row * 768 + d] = tp[d] + sp[d] + pe;
    }
}

__global__ __launch_bounds__(TB)
void layernorm_kernel(const float* __restrict__ in, const float* __restrict__ g,
                      const float* __restrict__ b, _Float16* __restrict__ outH,
                      float* __restrict__ outF, int D)
{
    __shared__ float red[TB];
    int row = blockIdx.x, tid = threadIdx.x;
    const float* p = in + (long)row * D;
    float s = 0.f;
    for (int d = tid; d < D; d += TB) s += p[d];
    red[tid] = s; __syncthreads();
    for (int o = TB / 2; o > 0; o >>= 1) { if (tid < o) red[tid] += red[tid + o]; __syncthreads(); }
    float mean = red[0] / D;
    __syncthreads();
    float v = 0.f;
    for (int d = tid; d < D; d += TB) { float q = p[d] - mean; v += q * q; }
    red[tid] = v; __syncthreads();
    for (int o = TB / 2; o > 0; o >>= 1) { if (tid < o) red[tid] += red[tid + o]; __syncthreads(); }
    float inv = rsqrtf(red[0] / D + 1e-5f);
    for (int d = tid; d < D; d += TB) {
        float y = (p[d] - mean) * inv * g[d] + b[d];
        outH[(long)row * D + d] = (_Float16)y;
        if (outF) outF[(long)row * D + d] = y;
    }
}

__global__ void transpose_f32_f16(const float* __restrict__ in, _Float16* __restrict__ out,
                                  int R, int C)
{
    __shared__ float tile[32][33];
    long zoff = (long)blockIdx.z * R * C;
    in  += zoff; out += zoff;
    int c0 = blockIdx.x * 32, r0 = blockIdx.y * 32;
    int tx = threadIdx.x, ty = threadIdx.y;
    for (int i = ty; i < 32; i += 8) {
        int r = r0 + i, c = c0 + tx;
        if (r < R && c < C) tile[i][tx] = in[(long)r * C + c];
    }
    __syncthreads();
    for (int i = ty; i < 32; i += 8) {
        int c = c0 + i, r = r0 + tx;
        if (r < R && c < C) out[(long)c * R + r] = (_Float16)tile[tx][i];
    }
}

__global__ __launch_bounds__(TB)
void softmax_mask_kernel(const float* __restrict__ scores,
                         const unsigned char* __restrict__ mask,
                         _Float16* __restrict__ attnH)
{
    __shared__ float red[TB];
    int row = blockIdx.x;
    int b   = row >> 11;
    int tid = threadIdx.x;
    const float* p = scores + (long)row * 512;
    const unsigned char* mrow = mask + (long)b * 512;
    float v0 = mrow[tid]       ? 1e-9f : p[tid];
    float v1 = mrow[tid + 256] ? 1e-9f : p[tid + 256];
    red[tid] = fmaxf(v0, v1); __syncthreads();
    for (int o = TB / 2; o > 0; o >>= 1) { if (tid < o) red[tid] = fmaxf(red[tid], red[tid + o]); __syncthreads(); }
    float mx = red[0];
    __syncthreads();
    float e0 = __expf(v0 - mx), e1 = __expf(v1 - mx);
    red[tid] = e0 + e1; __syncthreads();
    for (int o = TB / 2; o > 0; o >>= 1) { if (tid < o) red[tid] += red[tid + o]; __syncthreads(); }
    float rinv = 1.f / red[0];
    attnH[(long)row * 512 + tid]       = (_Float16)(e0 * rinv);
    attnH[(long)row * 512 + tid + 256] = (_Float16)(e1 * rinv);
}

__global__ __launch_bounds__(TB)
void log_softmax_kernel(float* __restrict__ logits, int V)
{
    __shared__ float red[TB];
    int row = blockIdx.x, tid = threadIdx.x;
    float* p = logits + (long)row * V;
    float mx = -3.4e38f;
    for (int i = tid; i < V; i += TB) mx = fmaxf(mx, p[i]);
    red[tid] = mx; __syncthreads();
    for (int o = TB / 2; o > 0; o >>= 1) { if (tid < o) red[tid] = fmaxf(red[tid], red[tid + o]); __syncthreads(); }
    mx = red[0];
    __syncthreads();
    float s = 0.f;
    for (int i = tid; i < V; i += TB) s += __expf(p[i] - mx);
    red[tid] = s; __syncthreads();
    for (int o = TB / 2; o > 0; o >>= 1) { if (tid < o) red[tid] += red[tid + o]; __syncthreads(); }
    float lse = mx + __logf(red[0]);
    for (int i = tid; i < V; i += TB) p[i] -= lse;
}

__global__ void cls_kernel(const float* __restrict__ enc, const float* __restrict__ Wc,
                           const float* __restrict__ bc, float* __restrict__ out)
{
    int o = threadIdx.x;
    if (o < 8) {
        int b = o >> 1, j = o & 1;
        const float* e = enc + (long)b * 512 * 768;
        float s = bc[j];
        for (int d = 0; d < 768; ++d) s += e[d] * Wc[d * 2 + j];
        out[o] = s;
    }
}

// ---------------------------------------------------------------------------
extern "C" void kernel_launch(void* const* d_in, const int* in_sizes, int n_in,
                              void* d_out, int out_size, void* d_ws, size_t ws_size,
                              hipStream_t stream)
{
    (void)in_sizes; (void)n_in; (void)out_size; (void)ws_size;

    const int*           x     = (const int*)d_in[0];
    const unsigned char* amask = (const unsigned char*)d_in[1];
    const float* tokE  = (const float*)d_in[2];
    const float* segE  = (const float*)d_in[3];
    const float* lnEg  = (const float*)d_in[4];
    const float* lnEb  = (const float*)d_in[5];
    const float* Wq    = (const float*)d_in[6];
    const float* bq    = (const float*)d_in[7];
    const float* Wk    = (const float*)d_in[8];
    const float* bk    = (const float*)d_in[9];
    const float* Wv    = (const float*)d_in[10];
    const float* bv    = (const float*)d_in[11];
    const float* Wo    = (const float*)d_in[12];
    const float* bo    = (const float*)d_in[13];
    const float* lnAg  = (const float*)d_in[14];
    const float* lnAb  = (const float*)d_in[15];
    const float* W1    = (const float*)d_in[16];
    const float* b1    = (const float*)d_in[17];
    const float* W2    = (const float*)d_in[18];
    const float* b2    = (const float*)d_in[19];
    const float* lnNg  = (const float*)d_in[20];
    const float* lnNb  = (const float*)d_in[21];
    const float* Wproj = (const float*)d_in[22];
    const float* bproj = (const float*)d_in[23];
    const float* Wcls  = (const float*)d_in[24];
    const float* bcls  = (const float*)d_in[25];

    const int Bsz = 4, T = 512, S = 2048, D = 768, H = 4, DH = 768, V = 32000, HD = 3072;

    char* wp = (char*)d_ws;
    auto alloc = [&](size_t bytes) -> void* {
        void* p = (void*)wp;
        wp += (bytes + 255) & ~(size_t)255;
        return p;
    };
    float*    emb    = (float*)   alloc((size_t)S * D * 4);
    float*    tmp    = (float*)   alloc((size_t)S * D * 4);
    float*    enc    = (float*)   alloc((size_t)S * D * 4);
    float*    scores = (float*)   alloc((size_t)Bsz * H * T * T * 4);
    _Float16* embH   = (_Float16*)alloc((size_t)S * D * 2);
    _Float16* WqT    = (_Float16*)alloc((size_t)HD * D * 2);
    _Float16* WkT    = (_Float16*)alloc((size_t)HD * D * 2);
    _Float16* WvT    = (_Float16*)alloc((size_t)HD * D * 2);
    _Float16* WoT    = (_Float16*)alloc((size_t)D * HD * 2);
    _Float16* W1T    = (_Float16*)alloc((size_t)D * D * 2);
    _Float16* W2T    = (_Float16*)alloc((size_t)D * D * 2);
    _Float16* WpT    = (_Float16*)alloc((size_t)V * D * 2);
    _Float16* Qh     = (_Float16*)alloc((size_t)S * HD * 2);
    _Float16* Kh     = (_Float16*)alloc((size_t)S * HD * 2);
    _Float16* Vt     = (_Float16*)alloc((size_t)HD * S * 2);
    _Float16* attnH  = (_Float16*)alloc((size_t)Bsz * H * T * T * 2);
    _Float16* catH   = (_Float16*)alloc((size_t)S * HD * 2);
    _Float16* aoH    = (_Float16*)alloc((size_t)S * D * 2);
    _Float16* hH     = (_Float16*)alloc((size_t)S * D * 2);
    _Float16* encH   = (_Float16*)alloc((size_t)S * D * 2);

    // batched (attention) GEMM path
    auto gemm = [&](const _Float16* A, const _Float16* B, const float* bias,
                    float* Cf, _Float16* Ch,
                    int M, int N, int K, int lda, int ldb, int ldc,
                    long sAb, long sAh, long sBb, long sBh, long sCb, long sCh,
                    int nb, int nh, float alpha, int act, int outTrans) {
        dim3 g(N / 64, M / 64, nb * nh);
        gemm_wmma_nt<<<g, dim3(TB), 0, stream>>>(A, B, bias, Cf, Ch, M, N, K,
                                                 lda, ldb, ldc, sAb, sAh, sBb, sBh,
                                                 sCb, sCh, nh, alpha, act, outTrans);
    };
    // large-GEMM TDM path (M%128==0, N%64==0, K%64==0)
    auto gemmT = [&](const _Float16* A, const _Float16* B, const float* bias,
                     float* Cf, _Float16* Ch, int M, int N, int K,
                     int lda, int ldb, int ldc, float alpha, int act, int outTrans) {
        dim3 g(N / 64, M / 128, 1);
        gemm_wmma_nt_tdm<<<g, dim3(TB), 0, stream>>>(A, B, bias, Cf, Ch, M, N, K,
                                                     lda, ldb, ldc, alpha, act, outTrans);
    };

    // 1) embedding + LN -> f16 operand
    embed_kernel<<<S, TB, 0, stream>>>(x, tokE, segE, emb);
    layernorm_kernel<<<S, TB, 0, stream>>>(emb, lnEg, lnEb, embH, nullptr, D);

    // 2) weight convert+transpose (f32 [R,C] -> f16 [C,R])
    dim3 tb(32, 8);
    transpose_f32_f16<<<dim3(24, 24, 4),   tb, 0, stream>>>(Wq,    WqT, 768,  768);
    transpose_f32_f16<<<dim3(24, 24, 4),   tb, 0, stream>>>(Wk,    WkT, 768,  768);
    transpose_f32_f16<<<dim3(24, 24, 4),   tb, 0, stream>>>(Wv,    WvT, 768,  768);
    transpose_f32_f16<<<dim3(24, 96, 1),   tb, 0, stream>>>(Wo,    WoT, 3072, 768);
    transpose_f32_f16<<<dim3(24, 24, 1),   tb, 0, stream>>>(W1,    W1T, 768,  768);
    transpose_f32_f16<<<dim3(24, 24, 1),   tb, 0, stream>>>(W2,    W2T, 768,  768);
    transpose_f32_f16<<<dim3(1000, 24, 1), tb, 0, stream>>>(Wproj, WpT, 768,  32000);

    // 3) QKV projections (TDM path; V pre-transposed for the NT ctx GEMM)
    gemmT(embH, WqT, bq, nullptr, Qh, S, HD, D, D, D, HD, 1.f, 0, 0);
    gemmT(embH, WkT, bk, nullptr, Kh, S, HD, D, D, D, HD, 1.f, 0, 0);
    gemmT(embH, WvT, bv, nullptr, Vt, S, HD, D, D, D, S,  1.f, 0, 1);

    // 4) scores = Q K^T * T^-0.5 (batched over b,h)
    gemm(Qh, Kh, nullptr, scores, nullptr,
         T, T, DH, HD, HD, T,
         (long)T * HD, (long)DH, (long)T * HD, (long)DH, (long)H * T * T, (long)T * T,
         Bsz, H, 0.044194173824159216f, 0, 0);

    // 5) mask(->1e-9) + softmax -> f16 attn
    softmax_mask_kernel<<<Bsz * H * T, TB, 0, stream>>>(scores, amask, attnH);

    // 6) ctx = attn @ V -> head-concat layout (f16)
    gemm(attnH, Vt, nullptr, nullptr, catH,
         T, DH, T, T, S, HD,
         (long)H * T * T, (long)T * T, (long)T, (long)DH * S, (long)T * HD, (long)DH,
         Bsz, H, 1.f, 0, 0);

    // 7) output projection + LN
    gemmT(catH, WoT, bo, tmp, nullptr, S, D, HD, HD, HD, D, 1.f, 0, 0);
    layernorm_kernel<<<S, TB, 0, stream>>>(tmp, lnAg, lnAb, aoH, nullptr, D);

    // 8) FFN (GELU fused into first GEMM epilogue)
    gemmT(aoH, W1T, b1, nullptr, hH, S, DH, D, D, D, DH, 1.f, 1, 0);
    gemmT(hH, W2T, b2, tmp, nullptr, S, D, DH, DH, DH, D, 1.f, 0, 0);
    layernorm_kernel<<<S, TB, 0, stream>>>(tmp, lnNg, lnNb, encH, enc, D);

    // 9) vocab projection straight into d_out, then in-place log-softmax
    float* logits = (float*)d_out;
    gemmT(encH, WpT, bproj, logits, nullptr, S, V, D, D, D, V, 1.f, 0, 0);
    log_softmax_kernel<<<S, TB, 0, stream>>>(logits, V);

    // 10) cls head -> tail of d_out
    cls_kernel<<<1, 32, 0, stream>>>(enc, Wcls, bcls, logits + (size_t)S * V);
}
